// ScoringHead_89395449299685
// MI455X (gfx1250) — matrix operand order repaired and checked by reading, not compiled, not run on hardware
//
#include <hip/hip_runtime.h>
#include <hip/hip_bf16.h>

// ---------------------------------------------------------------------------
// MI455X (gfx1250) implementation.
// Pipeline: proj (WMMA f16) -> l2norm -> score GEMM (WMMA f16, writes z=scores/2
// into d_out) -> rowwise entmax-1.5 via bisection on sum(max(z-tau,0)^2)=1.
// Memory floor: 3 passes over the 256MB score matrix ~= 33us @ 23.3 TB/s; all
// compute (8.6 GFLOP GEMM on wmma pipes + ~7.5 GFLOP VALU bisection) hides
// under that. Projection matrices (1MB f16 each) stay L2-resident (192MB L2),
// so the score GEMM streams B tiles straight from cache, no LDS staging.
// ---------------------------------------------------------------------------

typedef __attribute__((ext_vector_type(16))) _Float16 v16h;
typedef __attribute__((ext_vector_type(8)))  _Float16 v8h;
typedef __attribute__((ext_vector_type(8)))  float    v8f;

#define NROWS 8192   // S == E == 8192
#define HDIM  512
#define PDIM  64

// ---- WMMA operand loaders (contiguous k order, same for A and B) -------
// A (16x32 f16, M x K): lane L holds row m=L&15; half=L>=16 selects K groups
//   elem i -> K = i + 16*half  (16 contiguous halves)
// B (32x16 f16, K x N), sourced from row-major Bsrc (N x K, i.e. A*Bsrc^T):
//   lane L holds col n=L&15; elem i -> K = i + 16*half  (16 contiguous halves)

__device__ inline v16h load_a_cvt(const float* __restrict__ A, int lda,
                                  int m0, int k0, int lane) {
  const int m = m0 + (lane & 15);
  const int h = (lane >> 4) & 1;
  const float* row = A + (size_t)m * lda + k0 + 16 * h;
  v16h a;
#pragma unroll
  for (int i = 0; i < 8; ++i) a[i] = (_Float16)row[i];
#pragma unroll
  for (int i = 0; i < 8; ++i) a[8 + i] = (_Float16)row[8 + i];
  return a;
}

__device__ inline v16h load_bt_cvt(const float* __restrict__ B, int ldb,
                                   int n0, int k0, int lane) {
  const int n = n0 + (lane & 15);
  const int h = (lane >> 4) & 1;
  const float* row = B + (size_t)n * ldb + k0 + 16 * h;
  v16h b;
#pragma unroll
  for (int i = 0; i < 16; ++i) b[i] = (_Float16)row[i];
  return b;
}

__device__ inline v16h load_a_h16(const _Float16* __restrict__ A, int lda,
                                  int m0, int k0, int lane) {
  const int m = m0 + (lane & 15);
  const int h = (lane >> 4) & 1;
  const _Float16* row = A + (size_t)m * lda + k0 + 16 * h;
  v8h lo = *(const v8h*)(row);        // 16B aligned
  v8h hi = *(const v8h*)(row + 8);    // 16B aligned
  v16h a;
#pragma unroll
  for (int i = 0; i < 8; ++i) { a[i] = lo[i]; a[8 + i] = hi[i]; }
  return a;
}

__device__ inline v16h load_bt_h16(const _Float16* __restrict__ B, int ldb,
                                   int n0, int k0, int lane) {
  const int n = n0 + (lane & 15);
  const int h = (lane >> 4) & 1;
  // 32B-aligned: n*128B + {0,64}B + h*32B  -> single v16h (2 x b128) load
  return *(const v16h*)(B + (size_t)n * ldb + k0 + 16 * h);
}

// ---- 1) projection: raw[r][p] = dot(X[r,:], W[p,:]) + b[p] -----------------
// 8 waves/block; wave owns a 16-row strip x all 64 cols (4 accumulators).
__global__ __launch_bounds__(256) void proj_kernel(
    const float* __restrict__ X, const float* __restrict__ W,
    const float* __restrict__ bias, float* __restrict__ raw) {
  const int lane = threadIdx.x & 31;
  const int wave = threadIdx.x >> 5;
  const int m0 = blockIdx.x * 128 + wave * 16;

  v8f zero = {};
  v8f acc[4];
#pragma unroll
  for (int t = 0; t < 4; ++t) acc[t] = zero;

  for (int k0 = 0; k0 < HDIM; k0 += 32) {
    v16h a = load_a_cvt(X, HDIM, m0, k0, lane);
#pragma unroll
    for (int t = 0; t < 4; ++t) {
      v16h b = load_bt_cvt(W, HDIM, t * 16, k0, lane);
      acc[t] = __builtin_amdgcn_wmma_f32_16x16x32_f16(
          false, a, false, b, (short)0, acc[t], false, false);
    }
  }

  const int h  = (lane >> 4) & 1;
  const int nl = lane & 15;
#pragma unroll
  for (int t = 0; t < 4; ++t) {
    const int n  = t * 16 + nl;
    const float bv = bias[n];
#pragma unroll
    for (int r = 0; r < 8; ++r) {
      const int m = m0 + r + 8 * h;          // C layout: vgpr r -> M = r + 8*half
      raw[(size_t)m * PDIM + n] = acc[t][r] + bv;
    }
  }
}

// ---- 2) rowwise L2 normalize + f16 convert (one wave per row) --------------
__global__ __launch_bounds__(256) void norm_kernel(
    const float* __restrict__ raw, _Float16* __restrict__ outh) {
  const int lane = threadIdx.x & 31;
  const int wave = threadIdx.x >> 5;
  const int row = blockIdx.x * 8 + wave;
  const float v0 = raw[(size_t)row * PDIM + lane];
  const float v1 = raw[(size_t)row * PDIM + 32 + lane];
  float ss = v0 * v0 + v1 * v1;
#pragma unroll
  for (int off = 16; off > 0; off >>= 1) ss += __shfl_xor(ss, off, 32);
  const float inv = 1.0f / fmaxf(sqrtf(ss), 1e-12f);
  outh[(size_t)row * PDIM + lane]      = (_Float16)(v0 * inv);
  outh[(size_t)row * PDIM + 32 + lane] = (_Float16)(v1 * inv);
}

// ---- 3) score GEMM: z = 0.5*scale * (subP @ edgeP^T), written to d_out -----
// 128x128 tile per block; wave = 16 rows x 128 cols; K=64 -> 2 WMMA chunks.
__global__ __launch_bounds__(256) void score_kernel(
    const _Float16* __restrict__ Ah, const _Float16* __restrict__ Bh,
    const float* __restrict__ log_scale, float* __restrict__ out) {
  const int lane = threadIdx.x & 31;
  const int wave = threadIdx.x >> 5;
  const int m0 = blockIdx.y * 128 + wave * 16;
  const int n0 = blockIdx.x * 128;
  const float sc = 0.5f * fminf(fmaxf(expf(log_scale[0]), 0.5f), 20.0f);

  v16h a0 = load_a_h16(Ah, PDIM, m0, 0, lane);
  v16h a1 = load_a_h16(Ah, PDIM, m0, 32, lane);

  v8f zero = {};
  v8f acc[8];
#pragma unroll
  for (int t = 0; t < 8; ++t) acc[t] = zero;

#pragma unroll
  for (int t = 0; t < 8; ++t) {
    v16h b0 = load_bt_h16(Bh, PDIM, n0 + t * 16, 0, lane);
    acc[t] = __builtin_amdgcn_wmma_f32_16x16x32_f16(
        false, a0, false, b0, (short)0, acc[t], false, false);
    v16h b1 = load_bt_h16(Bh, PDIM, n0 + t * 16, 32, lane);
    acc[t] = __builtin_amdgcn_wmma_f32_16x16x32_f16(
        false, a1, false, b1, (short)0, acc[t], false, false);
  }

  const int h  = (lane >> 4) & 1;
  const int nl = lane & 15;
#pragma unroll
  for (int t = 0; t < 8; ++t) {
    const int n = n0 + t * 16 + nl;
#pragma unroll
    for (int r = 0; r < 8; ++r) {
      const int m = m0 + r + 8 * h;
      out[(size_t)m * NROWS + n] = acc[t][r] * sc;
    }
  }
}

// ---- 4) entmax-1.5 per row, in place.  tau solves sum(max(z-tau,0)^2)=1, ---
// tau in [zmax-1, zmax]; 32 bisection steps == exact sort-based tau to fp32.
__global__ __launch_bounds__(256) void entmax_kernel(float* __restrict__ zmat) {
  float* __restrict__ z = zmat + (size_t)blockIdx.x * NROWS;
  const int tid  = threadIdx.x;
  const int lane = tid & 31;
  const int wave = tid >> 5;
  __shared__ float part[8];

  float v[32];
#pragma unroll
  for (int j = 0; j < 32; ++j) v[j] = z[tid + j * 256];   // coalesced

  // block max
  float m = v[0];
#pragma unroll
  for (int j = 1; j < 32; ++j) m = fmaxf(m, v[j]);
#pragma unroll
  for (int off = 16; off > 0; off >>= 1) m = fmaxf(m, __shfl_xor(m, off, 32));
  if (lane == 0) part[wave] = m;
  __syncthreads();
  float zmax = part[0];
#pragma unroll
  for (int w = 1; w < 8; ++w) zmax = fmaxf(zmax, part[w]);
  __syncthreads();

  float lo = zmax - 1.0f, hi = zmax;   // f(lo)>=0, f(hi)=-1<0
#pragma unroll 1
  for (int it = 0; it < 32; ++it) {
    const float tau = 0.5f * (lo + hi);
    float s = 0.0f;
#pragma unroll
    for (int j = 0; j < 32; ++j) {
      const float d = fmaxf(v[j] - tau, 0.0f);
      s = fmaf(d, d, s);
    }
#pragma unroll
    for (int off = 16; off > 0; off >>= 1) s += __shfl_xor(s, off, 32);
    if (lane == 0) part[wave] = s;
    __syncthreads();
    float tot = 0.0f;
#pragma unroll
    for (int w = 0; w < 8; ++w) tot += part[w];   // same order everywhere
    __syncthreads();
    if (tot >= 1.0f) lo = tau; else hi = tau;     // uniform decision
  }

  const float tau = 0.5f * (lo + hi);
#pragma unroll
  for (int j = 0; j < 32; ++j) {
    const float d = fmaxf(v[j] - tau, 0.0f);
    z[tid + j * 256] = d * d;
  }
}

// ---------------------------------------------------------------------------
extern "C" void kernel_launch(void* const* d_in, const int* in_sizes, int n_in,
                              void* d_out, int out_size, void* d_ws, size_t ws_size,
                              hipStream_t stream) {
  const float* edge_repr = (const float*)d_in[0];
  const float* sub_repr  = (const float*)d_in[1];
  const float* W_sub     = (const float*)d_in[2];
  const float* b_sub     = (const float*)d_in[3];
  const float* W_edge    = (const float*)d_in[4];
  const float* b_edge    = (const float*)d_in[5];
  const float* log_scale = (const float*)d_in[6];
  float* out = (float*)d_out;

  const size_t rawBytes = (size_t)NROWS * PDIM * sizeof(float);   // 2 MB
  const size_t hBytes   = (size_t)NROWS * PDIM * sizeof(_Float16); // 1 MB
  char* ws = (char*)d_ws;
  float*    sub_raw  = (float*)(ws);
  float*    edge_raw = (float*)(ws + rawBytes);
  _Float16* sub_h    = (_Float16*)(ws + 2 * rawBytes);
  _Float16* edge_h   = (_Float16*)(ws + 2 * rawBytes + hBytes);

  proj_kernel<<<dim3(NROWS / 128), 256, 0, stream>>>(sub_repr,  W_sub,  b_sub,  sub_raw);
  proj_kernel<<<dim3(NROWS / 128), 256, 0, stream>>>(edge_repr, W_edge, b_edge, edge_raw);
  norm_kernel<<<dim3(NROWS / 8),   256, 0, stream>>>(sub_raw,  sub_h);
  norm_kernel<<<dim3(NROWS / 8),   256, 0, stream>>>(edge_raw, edge_h);
  score_kernel<<<dim3(NROWS / 128, NROWS / 128), 256, 0, stream>>>(sub_h, edge_h, log_scale, out);
  entmax_kernel<<<dim3(NROWS), 256, 0, stream>>>(out);
}
